// modelGamma_80152679678292
// MI455X (gfx1250) — matrix operand order repaired and checked by reading, not compiled, not run on hardware
//
#include <hip/hip_runtime.h>
#include <hip/hip_bf16.h>

#define N_ITER 3
#define N_GRID 2048
#define N_SENS 64
#define N_SNAP 128
#define BATCH  128
#define HID    32
#define VAR_NOISE 0.1f

typedef _Float16 half_t;
typedef __attribute__((ext_vector_type(16))) _Float16 v16h;
typedef __attribute__((ext_vector_type(8)))  _Float16 v8h;
typedef __attribute__((ext_vector_type(8)))  float    v8f;

// ---------------------------------------------------------------------------
// CDNA5 async global->LDS staging (ASYNCcnt path), with safe fallback
// ---------------------------------------------------------------------------
#if defined(__has_builtin)
#if __has_builtin(__builtin_amdgcn_global_load_async_to_lds_b128) && \
    __has_builtin(__builtin_amdgcn_s_wait_asynccnt)
#define HAS_ASYNC 1
#endif
#endif
#ifndef HAS_ASYNC
#define HAS_ASYNC 0
#endif

#if HAS_ASYNC
typedef int v4i_ __attribute__((vector_size(16)));
typedef __attribute__((address_space(1))) v4i_ gv4i;   // global (__device__) int4*
typedef __attribute__((address_space(3))) v4i_ lv4i;   // LDS (__shared__) int4*
#endif

__device__ __forceinline__ void cp16_g2l(void* ldst, const void* gsrc) {
#if HAS_ASYNC
  __builtin_amdgcn_global_load_async_to_lds_b128((gv4i*)gsrc, (lv4i*)ldst, 0, 0);
#else
  *(int4*)ldst = *(const int4*)gsrc;
#endif
}
__device__ __forceinline__ void async_join() {
#if HAS_ASYNC
  __builtin_amdgcn_s_wait_asynccnt(0);
#endif
  __syncthreads();
}

// ---------------------------------------------------------------------------
// WMMA fragment loaders (wave32, v_wmma_f32_16x16x32_f16 layouts, ISA 7.12.2)
//   A 16x32 f16 : row = lane&15 ; elems 0..7  -> K = kb + (lane>>4)*8 + i
//                                 elems 8..15 -> K = kb + 16 + (lane>>4)*8 + i
//   B 32x16 f16 : col = lane&15 ; same K mapping as A
//   C/D 16x16 f32: row = (lane>>4)*8 + r, col = lane&15
// ---------------------------------------------------------------------------
__device__ __forceinline__ v16h load_a_rm(const half_t* base, int row, int ld,
                                          int kb, int lane) {
  int kh = (lane >> 4) * 8;
  const half_t* p = base + row * ld + kb + kh;
  v8h lo = *(const v8h*)(p);       // 16B-aligned (kb%32==0, kh in {0,8})
  v8h hi = *(const v8h*)(p + 16);
  v16h a;
#pragma unroll
  for (int i = 0; i < 8; ++i) { a[i] = lo[i]; a[8 + i] = hi[i]; }
  return a;
}

// A-operand is M^T where M is stored row-major [K][ld]: elem(k) = M[k*ld + row]
__device__ __forceinline__ v16h load_a_tr(const half_t* base, int row, int ld,
                                          int kb, int lane) {
  int kh = (lane >> 4) * 8;
  v16h a;
#pragma unroll
  for (int i = 0; i < 8; ++i) {
    a[i]     = base[(kb + kh + i) * ld + row];
    a[8 + i] = base[(kb + 16 + kh + i) * ld + row];
  }
  return a;
}

__device__ __forceinline__ v16h load_b_rm(const half_t* base, int col, int ld,
                                          int kb, int lane) {
  int k0 = kb + (lane >> 4) * 8;
  v16h b;
#pragma unroll
  for (int i = 0; i < 8; ++i) {
    b[i]     = base[(k0 + i) * ld + col];
    b[8 + i] = base[(k0 + 16 + i) * ld + col];
  }
  return b;
}

#define WMMA_F16(a, b, c) \
  __builtin_amdgcn_wmma_f32_16x16x32_f16(false, (a), false, (b), (short)0, (c), false, false)

// ---------------------------------------------------------------------------
__global__ void init_ones(float* g, int n) {
  int i = blockIdx.x * blockDim.x + threadIdx.x;
  if (i < n) g[i] = 1.0f;
}

__global__ void f32_to_f16(const float* __restrict__ src,
                           half_t* __restrict__ dst, int n) {
  int i = blockIdx.x * blockDim.x + threadIdx.x;
  if (i < n) dst[i] = (half_t)src[i];
}

// ---------------------------------------------------------------------------
// cov[b] = A * diag(g_b) * A^T + VAR_NOISE*I     (64x64, K=2048, f16 WMMA)
// one block (128 thr = 4 waves) per batch; wave w owns output row-tile w
// ---------------------------------------------------------------------------
__global__ void build_cov(const half_t* __restrict__ Ah,
                          const half_t* __restrict__ gh,
                          float* __restrict__ cov) {
  int b    = blockIdx.x;
  int wave = threadIdx.x >> 5;
  int lane = threadIdx.x & 31;
  __shared__ __align__(16) half_t gs[N_GRID];
  for (int c = threadIdx.x; c < N_GRID / 8; c += 128)     // 4KB async stage
    cp16_g2l(gs + c * 8, gh + b * N_GRID + c * 8);
  async_join();

  int row   = wave * 16 + (lane & 15);
  int colLo = lane & 15;
  const half_t* pa = Ah + row * N_GRID;
  v8f zero = {};
  v8f acc[4];
#pragma unroll
  for (int nt = 0; nt < 4; ++nt) acc[nt] = zero;

  for (int kb = 0; kb < N_GRID; kb += 32) {
    __builtin_prefetch(pa + kb + 128, 0, 0);              // global_prefetch_b8
    int kh = (lane >> 4) * 8;
    v8h alo = *(const v8h*)(pa + kb + kh);
    v8h ahi = *(const v8h*)(pa + kb + 16 + kh);
    v8h glo = *(const v8h*)(gs + kb + kh);
    v8h ghi = *(const v8h*)(gs + kb + 16 + kh);
    v8h plo = alo * glo;
    v8h phi = ahi * ghi;
    v16h a;
#pragma unroll
    for (int i = 0; i < 8; ++i) { a[i] = plo[i]; a[8 + i] = phi[i]; }
    int k0 = kb + (lane >> 4) * 16;
#pragma unroll
    for (int nt = 0; nt < 4; ++nt) {
      int col = nt * 16 + colLo;
      v8h blo = *(const v8h*)(Ah + col * N_GRID + kb + kh);
      v8h bhi = *(const v8h*)(Ah + col * N_GRID + kb + 16 + kh);
      v16h bf;
#pragma unroll
      for (int i = 0; i < 8; ++i) { bf[i] = blo[i]; bf[8 + i] = bhi[i]; }
      acc[nt] = WMMA_F16(a, bf, acc[nt]);
    }
  }
  int rbase = (lane >> 4) * 8;
#pragma unroll
  for (int nt = 0; nt < 4; ++nt)
#pragma unroll
    for (int r = 0; r < 8; ++r) {
      int rr = wave * 16 + rbase + r;
      int cc = nt * 16 + colLo;
      cov[(b * 64 + rr) * 64 + cc] = acc[nt][r] + (rr == cc ? VAR_NOISE : 0.0f);
    }
}

// ---------------------------------------------------------------------------
// sinvh[b] = inverse(cov[b]) in f16 — Gauss-Jordan on SPD 64x64 in LDS
// ---------------------------------------------------------------------------
__global__ void invert64(const float* __restrict__ cov, half_t* __restrict__ sinvh) {
  int b = blockIdx.x, tid = threadIdx.x;
  __shared__ float M[64 * 128];
  __shared__ float colk[64];
  for (int i = 0; i < 64; ++i) {
    float v = (tid < 64) ? cov[(b * 64 + i) * 64 + tid]
                         : ((tid - 64 == i) ? 1.0f : 0.0f);
    M[i * 128 + tid] = v;
  }
  __syncthreads();
  for (int k = 0; k < 64; ++k) {
    if (tid < 64) colk[tid] = M[tid * 128 + k];
    __syncthreads();
    float invp = 1.0f / colk[k];
    M[k * 128 + tid] *= invp;
    __syncthreads();
    float mkj = M[k * 128 + tid];
    for (int i = 0; i < 64; ++i)
      if (i != k) M[i * 128 + tid] -= colk[i] * mkj;
    __syncthreads();
  }
  if (tid >= 64)
    for (int i = 0; i < 64; ++i)
      sinvh[(b * 64 + i) * 64 + (tid - 64)] = (half_t)M[i * 128 + tid];
}

// ---------------------------------------------------------------------------
// t1h[b] = sinv[b] @ data[b]   (64x128, K=64, f16 WMMA, async-staged LDS)
// ---------------------------------------------------------------------------
__global__ void solve_data(const half_t* __restrict__ sinvh,
                           const half_t* __restrict__ datah,
                           half_t* __restrict__ t1h) {
  int b    = blockIdx.x;
  int wave = threadIdx.x >> 5;
  int lane = threadIdx.x & 31;
  __shared__ __align__(16) half_t S[64 * 64];
  __shared__ __align__(16) half_t D[64 * 128];
  for (int c = threadIdx.x; c < 512; c += 128)            // 8KB
    cp16_g2l(S + c * 8, sinvh + b * 4096 + c * 8);
  for (int c = threadIdx.x; c < 1024; c += 128)           // 16KB
    cp16_g2l(D + c * 8, datah + b * 8192 + c * 8);
  async_join();

  int row = wave * 16 + (lane & 15);
  int colLo = lane & 15;
  v8f zero = {};
  for (int nt = 0; nt < 8; ++nt) {
    v8f acc = zero;
    for (int ks = 0; ks < 64; ks += 32) {
      v16h a  = load_a_rm(S, row, 64, ks, lane);
      v16h bf = load_b_rm(D, nt * 16 + colLo, 128, ks, lane);
      acc = WMMA_F16(a, bf, acc);
    }
    int rbase = wave * 16 + (lane >> 4) * 8;
    int cc    = nt * 16 + colLo;
#pragma unroll
    for (int r = 0; r < 8; ++r)
      t1h[(b * 64 + rbase + r) * 128 + cc] = (half_t)acc[r];
  }
}

// ---------------------------------------------------------------------------
// Fused T1/T2 per (batch, 64-gridpoint chunk):
//   T1[b,g] = mean_t ( (A[:,g]^T @ t1_b)[t] )^2   — WMMA + square-reduce
//   T2[b,g] = | A[:,g]^T @ sinv_b @ A[:,g] |      — WMMA + dot
// ---------------------------------------------------------------------------
__global__ void fused_T(const half_t* __restrict__ Ah,
                        const half_t* __restrict__ sinvh,
                        const half_t* __restrict__ t1h,
                        float* __restrict__ T1, float* __restrict__ T2) {
  int gc = blockIdx.x, b = blockIdx.y;
  int g0 = gc * 64;
  int tid = threadIdx.x, wave = tid >> 5, lane = tid & 31;
  __shared__ __align__(16) half_t S[64 * 64];     // sinv_b
  __shared__ __align__(16) half_t Tt[64 * 128];   // t1_b
  __shared__ __align__(16) half_t Ac[64 * 64];    // A[:, g0:g0+64]
  for (int c = tid; c < 512; c += 128)
    cp16_g2l(S + c * 8, sinvh + b * 4096 + c * 8);
  for (int c = tid; c < 1024; c += 128)
    cp16_g2l(Tt + c * 8, t1h + b * 8192 + c * 8);
  for (int c = tid; c < 512; c += 128) {
    int m = c >> 3, j8 = (c & 7) * 8;
    cp16_g2l(Ac + m * 64 + j8, Ah + m * N_GRID + g0 + j8);
  }
  async_join();

  int gl = wave * 16;
  int colLo = lane & 15;
  v8f zero = {};

  // ---- T1: U = Ac^T(16x64) @ t1(64x128); accumulate U^2 per row ----
  float sq[8];
#pragma unroll
  for (int r = 0; r < 8; ++r) sq[r] = 0.0f;
  for (int nt = 0; nt < 8; ++nt) {
    v8f acc = zero;
    for (int ks = 0; ks < 64; ks += 32) {
      v16h a  = load_a_tr(Ac, gl + colLo, 64, ks, lane);
      v16h bf = load_b_rm(Tt, nt * 16 + colLo, 128, ks, lane);
      acc = WMMA_F16(a, bf, acc);
    }
#pragma unroll
    for (int r = 0; r < 8; ++r) sq[r] += acc[r] * acc[r];
  }
#pragma unroll
  for (int r = 0; r < 8; ++r) {
    float v = sq[r];
    v += __shfl_xor(v, 1);
    v += __shfl_xor(v, 2);
    v += __shfl_xor(v, 4);
    v += __shfl_xor(v, 8);
    if (colLo == 0) {
      int grow = g0 + gl + (lane >> 4) * 8 + r;
      T1[b * N_GRID + grow] = v * (1.0f / (float)N_SNAP);
    }
  }

  // ---- T2: V = sinv(64x64) @ Ac[:,tile](64x16); dot columns with Ac ----
  float t2 = 0.0f;
  for (int mt = 0; mt < 4; ++mt) {
    v8f acc = zero;
    for (int ks = 0; ks < 64; ks += 32) {
      v16h a  = load_a_rm(S, mt * 16 + colLo, 64, ks, lane);
      v16h bf = load_b_rm(Ac, gl + colLo, 64, ks, lane);
      acc = WMMA_F16(a, bf, acc);
    }
#pragma unroll
    for (int r = 0; r < 8; ++r) {
      int mrow = mt * 16 + (lane >> 4) * 8 + r;
      t2 += acc[r] * (float)Ac[mrow * 64 + gl + colLo];
    }
  }
  t2 += __shfl_xor(t2, 16);
  if (lane < 16) T2[b * N_GRID + g0 + gl + colLo] = fabsf(t2);
}

// ---------------------------------------------------------------------------
// Per-gridpoint MLP: x=[T1,T2,g] -> relu(x W1+b1) -> relu(h W2+b2) -> h.W3+b3
// block = 128 threads (one per batch row) per gridpoint; layer2 on WMMA
// ---------------------------------------------------------------------------
__global__ void mlp(const float* __restrict__ T1, const float* __restrict__ T2,
                    const float* __restrict__ gin,
                    const float* __restrict__ W1, const float* __restrict__ b1,
                    const float* __restrict__ W2, const float* __restrict__ b2,
                    const float* __restrict__ W3, const float* __restrict__ b3,
                    float* __restrict__ gout) {
  int g = blockIdx.x;
  int tid = threadIdx.x, wave = tid >> 5, lane = tid & 31;
  __shared__ __align__(16) half_t H1[BATCH * HID];
  __shared__ __align__(16) half_t W2s[HID * HID];
  __shared__ float H2[BATCH * HID];

  float x0 = T1[tid * N_GRID + g];
  float x1 = T2[tid * N_GRID + g];
  float x2 = gin[tid * N_GRID + g];
  const float* w1  = W1 + g * 3 * HID;
  const float* bb1 = b1 + g * HID;
#pragma unroll
  for (int h = 0; h < HID; ++h) {
    float v = bb1[h] + x0 * w1[h] + x1 * w1[HID + h] + x2 * w1[2 * HID + h];
    H1[tid * HID + h] = (half_t)fmaxf(v, 0.0f);
  }
  for (int i = tid; i < HID * HID; i += 128) W2s[i] = (half_t)W2[g * HID * HID + i];
  __syncthreads();

  const float* bb2 = b2 + g * HID;
  int colLo = lane & 15;
  v8f zero = {};
#pragma unroll
  for (int mi = 0; mi < 2; ++mi) {
    int mt = wave * 2 + mi;
#pragma unroll
    for (int nt = 0; nt < 2; ++nt) {
      v16h a  = load_a_rm(H1, mt * 16 + colLo, HID, 0, lane);
      v16h bf = load_b_rm(W2s, nt * 16 + colLo, HID, 0, lane);
      v8f acc = WMMA_F16(a, bf, zero);
      int cc = nt * 16 + colLo;
      float bias = bb2[cc];
#pragma unroll
      for (int r = 0; r < 8; ++r) {
        int rr = mt * 16 + (lane >> 4) * 8 + r;
        H2[rr * HID + cc] = fmaxf(acc[r] + bias, 0.0f);
      }
    }
  }
  __syncthreads();

  const float* w3 = W3 + g * HID;
  float o = b3[g];
#pragma unroll
  for (int h = 0; h < HID; ++h) o += H2[tid * HID + h] * w3[h];
  gout[tid * N_GRID + g] = o;
}

// ---------------------------------------------------------------------------
extern "C" void kernel_launch(void* const* d_in, const int* in_sizes, int n_in,
                              void* d_out, int out_size, void* d_ws, size_t ws_size,
                              hipStream_t stream) {
  const float* data = (const float*)d_in[0];   // [128,64,128]
  const float* A    = (const float*)d_in[1];   // [64,2048]
  const float* W1   = (const float*)d_in[2];   // [3,2048,3,32]
  const float* b1   = (const float*)d_in[3];   // [3,2048,32]
  const float* W2   = (const float*)d_in[4];   // [3,2048,32,32]
  const float* b2   = (const float*)d_in[5];   // [3,2048,32]
  const float* W3   = (const float*)d_in[6];   // [3,2048,32]
  const float* b3   = (const float*)d_in[7];   // [3,2048]
  float* out = (float*)d_out;

  const int NG = BATCH * N_GRID;               // 262144
  float*  gA    = (float*)d_ws;
  float*  gB    = gA + NG;
  float*  cov   = gB + NG;                     // 128*64*64
  float*  T1b   = cov + BATCH * 64 * 64;
  float*  T2b   = T1b + NG;
  half_t* Ah    = (half_t*)(T2b + NG);         // 64*2048
  half_t* datah = Ah + N_SENS * N_GRID;        // 128*64*128
  half_t* gh    = datah + BATCH * 64 * 128;    // 128*2048
  half_t* sinvh = gh + NG;                     // 128*64*64
  half_t* t1h   = sinvh + BATCH * 64 * 64;     // 128*64*128   (~11.8 MB total)

  init_ones<<<(NG + 255) / 256, 256, 0, stream>>>(gA, NG);
  int nA = N_SENS * N_GRID, nD = BATCH * 64 * 128;
  f32_to_f16<<<(nA + 255) / 256, 256, 0, stream>>>(A, Ah, nA);
  f32_to_f16<<<(nD + 255) / 256, 256, 0, stream>>>(data, datah, nD);

  float* gcur = gA;
  float* gnext = gB;
  for (int it = 0; it < N_ITER; ++it) {
    f32_to_f16<<<(NG + 255) / 256, 256, 0, stream>>>(gcur, gh, NG);
    build_cov<<<BATCH, 128, 0, stream>>>(Ah, gh, cov);
    invert64<<<BATCH, 128, 0, stream>>>(cov, sinvh);
    solve_data<<<BATCH, 128, 0, stream>>>(sinvh, datah, t1h);
    fused_T<<<dim3(N_GRID / 64, BATCH), 128, 0, stream>>>(Ah, sinvh, t1h, T1b, T2b);
    float* gout = (it == N_ITER - 1) ? out : gnext;
    mlp<<<N_GRID, 128, 0, stream>>>(T1b, T2b, gcur,
                                    W1 + (size_t)it * N_GRID * 3 * HID,
                                    b1 + (size_t)it * N_GRID * HID,
                                    W2 + (size_t)it * N_GRID * HID * HID,
                                    b2 + (size_t)it * N_GRID * HID,
                                    W3 + (size_t)it * N_GRID * HID,
                                    b3 + (size_t)it * N_GRID,
                                    gout);
    float* tmp = gcur; gcur = gnext; gnext = tmp;
  }
}
